// Decoder_16037407883646
// MI455X (gfx1250) — matrix operand order repaired and checked by reading, not compiled, not run on hardware
//
#include <hip/hip_runtime.h>

// ---------------- problem dims ----------------
constexpr int cB = 32, cTENC = 512, cTMEL = 400;
constexpr int cA = 512, cM = 128, cF = 32, cKC = 31, cPAD = 15;
constexpr int cH3 = 1536;          // 3*H
constexpr int cXI = 768;           // P+E  (attn GRU input dim)
constexpr int cX2 = 1024;          // H+A  (out GRU input / att_h dim)

// ---------------- launch shape ----------------
constexpr int NBLK = 32;
constexpr int NTHR = 512;
constexpr int WPB = NTHR / 32;          // 16 waves / block (wave32)
constexpr int NWAVES = NBLK * WPB;      // 512
constexpr int NTHREADS = NBLK * NTHR;   // 16384

typedef unsigned short u16;
typedef __attribute__((ext_vector_type(16))) __bf16 v16bf;
typedef __attribute__((ext_vector_type(8)))  float  v8f;
typedef __attribute__((ext_vector_type(4)))  unsigned u32x4;
typedef __attribute__((ext_vector_type(4)))  float    f32x4;
typedef __attribute__((ext_vector_type(4)))  int      i32x4;
typedef __attribute__((ext_vector_type(8)))  int      i32x8;

// explicit global-address-space pointers: force GLOBAL_* (not FLAT_*) memory ops
#define ASG __attribute__((address_space(1)))
typedef ASG const float*  gcf;
typedef ASG float*        gf;
typedef ASG const u16*    gcu;
typedef ASG u16*          gu;
typedef ASG const int*    gci;
typedef ASG const u32x4*  gcq;
typedef ASG u32x4*        gq;
typedef ASG const f32x4*  gcf4;

union Frag { v16bf v; u16 h[16]; unsigned u[8]; u32x4 q[2]; };

__device__ __forceinline__ u16 f2bf(float x) {
  unsigned u = __float_as_uint(x);
  u += 0x7FFFu + ((u >> 16) & 1u);          // round-to-nearest-even
  return (u16)(u >> 16);
}
__device__ __forceinline__ float bf2f(u16 h) { return __uint_as_float(((unsigned)h) << 16); }
__device__ __forceinline__ float sigm(float x) { return 1.f / (1.f + __expf(-x)); }
__device__ __forceinline__ float fast_tanh(float x) {
#if __has_builtin(__builtin_amdgcn_tanhf)
  return __builtin_amdgcn_tanhf(x);          // v_tanh_f32 (CDNA5 TRANS op)
#else
  return tanhf(x);
#endif
}

// ---- WMMA fragment loaders (ISA 7.12.2 layouts, wave32), 2x b128 each (loadA_f32: 4x) ----
__device__ __forceinline__ v16bf loadA(gcu base, int ld, int lane) {
  int half = lane >> 4, m = lane & 15;
  gcq rp = (gcq)(base + m * ld);
  Frag f;
  f.q[0] = rp[half];        // K 0..7 (lo) / 8..15 (hi)
  f.q[1] = rp[2 + half];    // K 16..23 (lo) / 24..31 (hi)
  return f.v;
}
__device__ __forceinline__ v16bf loadB(gcu base, int ld, int lane) {
  int half = lane >> 4, n = lane & 15;
  gcq rp = (gcq)(base + n * ld);
  Frag f;
  f.q[0] = rp[half];
  f.q[1] = rp[2 + half];
  return f.v;
}
__device__ __forceinline__ v16bf loadB_lds(const u16* base, int ld, int lane) {
  int half = lane >> 4, n = lane & 15;
  const u32x4* rp = (const u32x4*)(base + n * ld);
  Frag f;
  f.q[0] = rp[half];
  f.q[1] = rp[2 + half];
  return f.v;
}
__device__ __forceinline__ v16bf loadA_f32(gcf base, int ld, int lane) {
  int half = lane >> 4, m = lane & 15;
  gcf4 r0 = (gcf4)(base + m * ld + 8 * half);
  gcf4 r1 = (gcf4)(base + m * ld + 16 + 8 * half);
  f32x4 x0 = r0[0], x1 = r0[1], x2 = r1[0], x3 = r1[1];
  Frag f;
#pragma unroll
  for (int i = 0; i < 4; ++i) {
    f.h[i] = f2bf(x0[i]); f.h[4 + i] = f2bf(x1[i]);
    f.h[8 + i] = f2bf(x2[i]); f.h[12 + i] = f2bf(x3[i]);
  }
  return f.v;
}

// ---------------- TDM: 1-D tensor load global -> LDS (D# per ISA 8.3/8.4) ----------------
// 6-arg toolchain form: (u32x4 g0, i32x8 g1, i32x4 g2, i32x4 g3, i32x8 extra, i32 cpol)
__device__ __forceinline__ void tdm_load_1d(unsigned lds_off, unsigned long long ga,
                                            unsigned nelem, unsigned dsz_log) {
#if __has_builtin(__builtin_amdgcn_tensor_load_to_lds)
  u32x4 g0;
  g0[0] = 1u;                                              // count=1 (user mode)
  g0[1] = lds_off;                                         // lds_addr
  g0[2] = (unsigned)(ga & 0xFFFFFFFFu);                    // global_addr lo
  g0[3] = (unsigned)((ga >> 32) & 0x01FFFFFFu) | (2u << 30); // global_addr hi | type=2
  i32x8 g1;
  g1[0] = (int)(dsz_log << 16);                            // wgmask=0, data_size
  g1[1] = (int)((nelem & 0xFFFFu) << 16);                  // tensor_dim0[15:0]
  g1[2] = (int)(((nelem >> 16) & 0xFFFFu) | (1u << 16));   // tensor_dim0[31:16], tensor_dim1=1
  g1[3] = (int)((nelem & 0xFFFFu) << 16);                  // tile_dim0 = nelem
  g1[4] = 0;                                               // tile_dim1/2 unused
  g1[5] = (int)nelem;                                      // tensor_dim0_stride lo
  g1[6] = 0; g1[7] = 0;
  i32x4 z4 = {};
  i32x8 z8 = {};
  __builtin_amdgcn_tensor_load_to_lds(g0, g1, z4, z4, z8, 0);
#else
  (void)lds_off; (void)ga; (void)nelem; (void)dsz_log;
#endif
}
__device__ __forceinline__ unsigned lds_offset_of(const void* p) {
  return (unsigned)(unsigned long long)(__attribute__((address_space(3))) const void*)p;
}

// ---------------- kernel argument block (host side: plain pointers) ----------------
struct Args {
  const float *enc, *pre_w1, *pre_w2, *attn_wi, *attn_wh, *attn_bi, *attn_bh;
  const float *q_w, *k_w, *score_w, *score_b, *loc_conv, *loc_w, *loc_b;
  const float *out_wi, *out_wh, *out_bi, *out_bh, *dec_w, *dec_b, *gate_w, *gate_b;
  const int *text_lens, *mel_lens;
  float *o_mel, *o_gate, *o_attn, *o_mask;
  unsigned *bar;  // [0]=count [1]=generation
  u16 *attn_wi_t, *attn_wh_t, *out_wi_t, *out_wh_t, *q_w_t, *k_w_t, *loc_w_t;
  u16 *pre_w1_t, *pre_w2_t, *dec_w_t;
  u16 *keys;      // [B*TENC][A] bf16
  u16 *keys_t;    // [B][A][TENC] bf16 (score pass: contiguous in t)
  u16 *loc_f;     // [B*TENC][F] bf16 (per-step conv output, A-tile-ready layout)
  u16 *xcat, *xcat2, *xcat3, *h1b, *melb;
  float *h_attn, *h_out, *attw, *attcum, *qb, *sb, *gi, *gh;
};

// device-side view with explicit global address space
struct GA {
  gcf enc, pre_w1, pre_w2, attn_wi, attn_wh, attn_bi, attn_bh;
  gcf q_w, k_w, score_w, score_b, loc_conv, loc_w, loc_b;
  gcf out_wi, out_wh, out_bi, out_bh, dec_w, dec_b, gate_w, gate_b;
  gci text_lens, mel_lens;
  gf o_mel, o_gate, o_attn, o_mask;
  gu attn_wi_t, attn_wh_t, out_wi_t, out_wh_t, q_w_t, k_w_t, loc_w_t;
  gu pre_w1_t, pre_w2_t, dec_w_t;
  gu keys, keys_t, loc_f, xcat, xcat2, xcat3, h1b, melb;
  gf h_attn, h_out, attw, attcum, qb, sb, gi, gh;
};

// ---------------- device-wide barrier (32 resident blocks) ----------------
__device__ __forceinline__ void gsync(unsigned* bar) {
  __threadfence();
  __syncthreads();
  if (threadIdx.x == 0) {
    unsigned g = __hip_atomic_load(&bar[1], __ATOMIC_RELAXED, __HIP_MEMORY_SCOPE_AGENT);
    unsigned a = __hip_atomic_fetch_add(&bar[0], 1u, __ATOMIC_ACQ_REL, __HIP_MEMORY_SCOPE_AGENT);
    if (a == NBLK - 1) {
      __hip_atomic_store(&bar[0], 0u, __ATOMIC_RELAXED, __HIP_MEMORY_SCOPE_AGENT);
      __hip_atomic_fetch_add(&bar[1], 1u, __ATOMIC_ACQ_REL, __HIP_MEMORY_SCOPE_AGENT);
    } else {
      while (__hip_atomic_load(&bar[1], __ATOMIC_ACQUIRE, __HIP_MEMORY_SCOPE_AGENT) == g)
        __builtin_amdgcn_s_sleep(2);
    }
  }
  __syncthreads();
  __threadfence();
}

// ---------------- generic bf16 WMMA GEMM: Out[MrxN] = A[MrxK] @ Wt[NxK]^T (+bias) ----------------
__device__ void gemm_bf(gcu Ab, int lda, gcu Bt, int ldb,
                        gcf bias, gf Of, int ldf, gu Ob, int ldob,
                        int Mr, int N, int Kd, bool relu, int gw, int lane) {
  int nt = N >> 4, ks = Kd >> 5, tiles = (Mr >> 4) * nt;
  for (int t = gw; t < tiles; t += NWAVES) {
    int tm = t / nt, tn = t - tm * nt;
    gcu Abase = Ab + (tm * 16) * lda;
    gcu Bbase = Bt + (tn * 16) * ldb;
    v8f acc = {};
    for (int k = 0; k < ks; ++k) {
      v16bf a = loadA(Abase + 32 * k, lda, lane);
      v16bf b = loadB(Bbase + 32 * k, ldb, lane);
      acc = __builtin_amdgcn_wmma_f32_16x16x32_bf16(false, a, false, b, (short)0, acc, false, false);
    }
    int n = lane & 15, mh = (lane >> 4) * 8;
    float bv = bias ? bias[tn * 16 + n] : 0.f;
#pragma unroll
    for (int r = 0; r < 8; ++r) {
      float v = acc[r] + bv;
      if (relu) v = fmaxf(v, 0.f);
      int row = tm * 16 + mh + r, col = tn * 16 + n;
      if (Of) Of[row * ldf + col] = v;
      if (Ob) Ob[row * ldob + col] = f2bf(v);
    }
  }
}

// keys = enc(fp32 -> bf16 on the fly) @ k_w ; writes row-major keys AND t-contiguous keys_t
__device__ void gemm_keys(gcf Af, int lda, gcu Bt, int ldb,
                          gu keys, gu keys_t, int gw, int lane) {
  constexpr int N = cA, Kd = 1024, Mr = cB * cTENC;
  int nt = N >> 4, ks = Kd >> 5, tiles = (Mr >> 4) * nt;
  for (int t = gw; t < tiles; t += NWAVES) {
    int tm = t / nt, tn = t - tm * nt;
    gcf Abase = Af + (size_t)(tm * 16) * lda;
    gcu Bbase = Bt + (tn * 16) * ldb;
    v8f acc = {};
    for (int k = 0; k < ks; ++k) {
      v16bf a = loadA_f32(Abase + 32 * k, lda, lane);
      v16bf b = loadB(Bbase + 32 * k, ldb, lane);
      acc = __builtin_amdgcn_wmma_f32_16x16x32_bf16(false, a, false, b, (short)0, acc, false, false);
    }
    int n = lane & 15, mh = (lane >> 4) * 8;
#pragma unroll
    for (int r = 0; r < 8; ++r) {
      int row = tm * 16 + mh + r;            // = b*TENC + t
      int col = tn * 16 + n;                 // = a
      u16 hv = f2bf(acc[r]);
      keys[(size_t)row * cA + col] = hv;
      int b = row >> 9, tt = row & 511;
      keys_t[((size_t)b * cA + col) * cTENC + tt] = hv;
    }
  }
}

// transpose-convert weight W[K][N] fp32 -> Wt[N][K] bf16
__device__ void conv_t(gcf w, gu wt, int Kd, int N, int gtid) {
  for (int i = gtid; i < Kd * N; i += NTHREADS) {
    int n = i / Kd, k = i - n * Kd;
    wt[i] = f2bf(w[k * N + n]);
  }
}

// fused GRU pointwise: h = (1-z)*n + z*h ; also writes bf16 h into concat slot
__device__ void gru_ew(gcf gi, gcf gh, gf h, gu xc, int ldx, int gtid) {
  for (int i = gtid; i < cB * 512; i += NTHREADS) {
    int b = i >> 9, j = i & 511;
    gcf gib = gi + b * cH3;
    gcf ghb = gh + b * cH3;
    float r = sigm(gib[j] + ghb[j]);
    float z = sigm(gib[512 + j] + ghb[512 + j]);
    float n = fast_tanh(gib[1024 + j] + r * ghb[1024 + j]);
    float hv = (1.f - z) * n + z * h[i];
    h[i] = hv;
    xc[b * ldx + j] = f2bf(hv);
  }
}

// location conv: one thread per (b,t); weights from LDS; output rows are WMMA-A-ready
__device__ void conv_phase(const GA& A, const float* sc, int gtid) {
  int b = gtid >> 9, t = gtid & 511;
  gcf aw = A.attw + b * cTENC;
  gcf ac = A.attcum + b * cTENC;
  float acc[cF];
#pragma unroll
  for (int f = 0; f < cF; ++f) acc[f] = 0.f;
  for (int k = 0; k < cKC; ++k) {
    int tt = t + k - cPAD;
    bool ok = (tt >= 0) && (tt < cTENC);
    float a0 = ok ? aw[tt] : 0.f;
    float c0 = ok ? ac[tt] : 0.f;
#pragma unroll
    for (int f = 0; f < cF; ++f)
      acc[f] += sc[f * 62 + k] * a0 + sc[f * 62 + cKC + k] * c0;
  }
  union { u16 h[cF]; u32x4 q[4]; } o;
#pragma unroll
  for (int f = 0; f < cF; ++f) o.h[f] = f2bf(acc[f]);
  gq dst = (gq)(A.loc_f + (size_t)gtid * cF);
#pragma unroll
  for (int i = 0; i < 4; ++i) dst[i] = o.q[i];
}

// fused loc-projection (A from loc_f, B=loc_w from LDS) -> tanh-score
__device__ void score_phase(const GA& A, const u16* slocw, int gw, int lane) {
  int n = lane & 15, mh = (lane >> 4) * 8;
  float sb0 = A.score_b[0];
  for (int tile = gw; tile < cB * (cTENC / 16); tile += NWAVES) {
    int b = tile >> 5;
    int t0 = (tile & 31) << 4;
    v16bf af = loadA(A.loc_f + (size_t)(b * cTENC + t0) * cF, cF, lane);
    float part[8] = {0, 0, 0, 0, 0, 0, 0, 0};
    for (int tn = 0; tn < cA / 16; ++tn) {
      v16bf bb = loadB_lds(slocw + (tn * 16) * cF, cF, lane);
      v8f acc = {};
      acc = __builtin_amdgcn_wmma_f32_16x16x32_bf16(false, af, false, bb, (short)0, acc, false, false);
      int a0 = tn * 16 + n;
      float addc = A.qb[b * cA + a0] + A.loc_b[a0];
      float sw = A.score_w[a0];
      Frag kf;  // 8 keys (rows t0+mh..+7, col a0): ONE aligned b128 load from keys_t
      kf.q[0] = *((gcq)(A.keys_t + ((size_t)(b * cA + a0)) * cTENC + t0 + mh));
#pragma unroll
      for (int r = 0; r < 8; ++r)
        part[r] += fast_tanh(acc[r] + addc + bf2f(kf.h[r])) * sw;
    }
    for (int off = 1; off < 16; off <<= 1) {
#pragma unroll
      for (int r = 0; r < 8; ++r) part[r] += __shfl_xor(part[r], off, 32);
    }
    if (n == 0) {
      int tl = A.text_lens[b];
      for (int r = 0; r < 8; ++r) {
        int tt = t0 + mh + r;
        float sv = part[r] + sb0;
        if (tt > tl) sv = -__builtin_inff();
        A.sb[b * cTENC + tt] = sv;
      }
    }
  }
}

// per-block-b softmax + context; writes attn output and ctx bf16 into all three concat buffers
__device__ void softmax_ctx(const GA& A, int step) {
  __shared__ float red[NTHR];
  __shared__ float wsh[cTENC];
  int b = blockIdx.x, tid = threadIdx.x;
  float v = A.sb[b * cTENC + tid];
  red[tid] = v; __syncthreads();
  for (int o = NTHR / 2; o > 0; o >>= 1) { if (tid < o) red[tid] = fmaxf(red[tid], red[tid + o]); __syncthreads(); }
  float mx = red[0]; __syncthreads();
  float e = __expf(v - mx);            // exp(-inf)=0 for masked slots
  red[tid] = e; __syncthreads();
  for (int o = NTHR / 2; o > 0; o >>= 1) { if (tid < o) red[tid] += red[tid + o]; __syncthreads(); }
  float w = e / red[0];
  wsh[tid] = w;
  A.attw[b * cTENC + tid] = w;
  A.attcum[b * cTENC + tid] += w;
  A.o_attn[((size_t)b * cTMEL + step) * cTENC + tid] = w;
  __syncthreads();
  float acc = 0.f;
  gcu kb = A.keys + (size_t)b * cTENC * cA;
  for (int t = 0; t < cTENC; ++t) {
    if ((t & 31) == 0 && t + 32 < cTENC)
      __builtin_prefetch((const void*)(kb + (size_t)(t + 32) * cA + tid), 0, 1); // global_prefetch_b8
    acc += wsh[t] * bf2f(kb[(size_t)t * cA + tid]);
  }
  u16 cbv = f2bf(acc);
  A.xcat[b * cXI + 256 + tid] = cbv;   // [dec_in | ctx]
  A.xcat2[b * cX2 + 512 + tid] = cbv;  // [h_attn | ctx]
  A.xcat3[b * cX2 + 512 + tid] = cbv;  // [h_out  | ctx]
}

// mel GEMM with fused output masking + melb, plus stop-gate and mask (one phase, one barrier)
__device__ void mel_phase(const GA& A, int step, int gw, int lane) {
  constexpr int nt = cM >> 4, ks = cX2 >> 5;
  int tiles = (cB >> 4) * nt;                       // 16
  for (int t = gw; t < tiles; t += NWAVES) {
    int tm = t / nt, tn = t - tm * nt;
    gcu Abase = A.xcat3 + (tm * 16) * cX2;
    gcu Bbase = A.dec_w_t + (tn * 16) * cX2;
    v8f acc = {};
    for (int k = 0; k < ks; ++k) {
      v16bf a = loadA(Abase + 32 * k, cX2, lane);
      v16bf b = loadB(Bbase + 32 * k, cX2, lane);
      acc = __builtin_amdgcn_wmma_f32_16x16x32_bf16(false, a, false, b, (short)0, acc, false, false);
    }
    int n = lane & 15, mh = (lane >> 4) * 8;
    float bv = A.dec_b[tn * 16 + n];
#pragma unroll
    for (int r = 0; r < 8; ++r) {
      int b = tm * 16 + mh + r, col = tn * 16 + n;
      float v = acc[r] + bv;
      A.melb[b * cM + col] = f2bf(v);
      bool msk = step > A.mel_lens[b];
      A.o_mel[((size_t)b * cTMEL + step) * cM + col] = msk ? 0.f : v;
    }
  }
  int gv = gw - 64;                                 // waves 64..95: stop gate + mask
  if (gv >= 0 && gv < cB) {
    int b = gv;
    float sum = 0.f;
    for (int j = lane; j < cX2; j += 32) sum += bf2f(A.xcat3[b * cX2 + j]) * A.gate_w[j];
    for (int off = 16; off >= 1; off >>= 1) sum += __shfl_xor(sum, off, 32);
    if (lane == 0) {
      bool msk = step > A.mel_lens[b];
      A.o_gate[b * cTMEL + step] = msk ? 1000.f : (sum + A.gate_b[0]);
      A.o_mask[b * cTMEL + step] = msk ? 1.f : 0.f;
    }
  }
}

// ---------------- persistent decoder ----------------
__global__ __launch_bounds__(NTHR, 1) void tacotron2_decoder_persistent(Args P) {
  int lane = threadIdx.x & 31;
  int gw = blockIdx.x * WPB + (threadIdx.x >> 5);
  int gtid = blockIdx.x * NTHR + threadIdx.x;

  GA A;
  A.enc = (gcf)P.enc; A.pre_w1 = (gcf)P.pre_w1; A.pre_w2 = (gcf)P.pre_w2;
  A.attn_wi = (gcf)P.attn_wi; A.attn_wh = (gcf)P.attn_wh;
  A.attn_bi = (gcf)P.attn_bi; A.attn_bh = (gcf)P.attn_bh;
  A.q_w = (gcf)P.q_w; A.k_w = (gcf)P.k_w;
  A.score_w = (gcf)P.score_w; A.score_b = (gcf)P.score_b;
  A.loc_conv = (gcf)P.loc_conv; A.loc_w = (gcf)P.loc_w; A.loc_b = (gcf)P.loc_b;
  A.out_wi = (gcf)P.out_wi; A.out_wh = (gcf)P.out_wh;
  A.out_bi = (gcf)P.out_bi; A.out_bh = (gcf)P.out_bh;
  A.dec_w = (gcf)P.dec_w; A.dec_b = (gcf)P.dec_b;
  A.gate_w = (gcf)P.gate_w; A.gate_b = (gcf)P.gate_b;
  A.text_lens = (gci)P.text_lens; A.mel_lens = (gci)P.mel_lens;
  A.o_mel = (gf)P.o_mel; A.o_gate = (gf)P.o_gate; A.o_attn = (gf)P.o_attn; A.o_mask = (gf)P.o_mask;
  A.attn_wi_t = (gu)P.attn_wi_t; A.attn_wh_t = (gu)P.attn_wh_t;
  A.out_wi_t = (gu)P.out_wi_t; A.out_wh_t = (gu)P.out_wh_t;
  A.q_w_t = (gu)P.q_w_t; A.k_w_t = (gu)P.k_w_t; A.loc_w_t = (gu)P.loc_w_t;
  A.pre_w1_t = (gu)P.pre_w1_t; A.pre_w2_t = (gu)P.pre_w2_t; A.dec_w_t = (gu)P.dec_w_t;
  A.keys = (gu)P.keys; A.keys_t = (gu)P.keys_t; A.loc_f = (gu)P.loc_f;
  A.xcat = (gu)P.xcat; A.xcat2 = (gu)P.xcat2; A.xcat3 = (gu)P.xcat3;
  A.h1b = (gu)P.h1b; A.melb = (gu)P.melb;
  A.h_attn = (gf)P.h_attn; A.h_out = (gf)P.h_out;
  A.attw = (gf)P.attw; A.attcum = (gf)P.attcum;
  A.qb = (gf)P.qb; A.sb = (gf)P.sb; A.gi = (gf)P.gi; A.gh = (gf)P.gh;

  __shared__ u16   slocw[cA * cF];        // transposed loc_w, 32 KB bf16
  __shared__ float sconv[cF * 2 * cKC];   // raw conv taps, 7.9 KB f32

  // phase 0: zero recurrent state, convert all weights to transposed bf16
  for (int i = gtid; i < cB * 512; i += NTHREADS) { A.h_attn[i] = 0.f; A.h_out[i] = 0.f; }
  for (int i = gtid; i < cB * cTENC; i += NTHREADS) { A.attw[i] = 0.f; A.attcum[i] = 0.f; }
  for (int i = gtid; i < cB * cXI; i += NTHREADS) A.xcat[i] = 0;
  for (int i = gtid; i < cB * cX2; i += NTHREADS) { A.xcat2[i] = 0; A.xcat3[i] = 0; }
  conv_t(A.attn_wi, A.attn_wi_t, cXI, cH3, gtid);
  conv_t(A.attn_wh, A.attn_wh_t, 512, cH3, gtid);
  conv_t(A.out_wi, A.out_wi_t, cX2, cH3, gtid);
  conv_t(A.out_wh, A.out_wh_t, 512, cH3, gtid);
  conv_t(A.q_w, A.q_w_t, 512, cA, gtid);
  conv_t(A.k_w, A.k_w_t, 1024, cA, gtid);
  conv_t(A.loc_w, A.loc_w_t, cF, cA, gtid);
  conv_t(A.pre_w1, A.pre_w1_t, 128, 256, gtid);
  conv_t(A.pre_w2, A.pre_w2_t, 256, 256, gtid);
  conv_t(A.dec_w, A.dec_w_t, cX2, cM, gtid);
  gsync(P.bar);

  // stage loc_w_t (32 KB bf16) and loc_conv (7.9 KB f32) into LDS via the Tensor Data Mover
#if __has_builtin(__builtin_amdgcn_tensor_load_to_lds)
  if (threadIdx.x < 32) {        // wave 0 issues both TDM descriptors
    tdm_load_1d(lds_offset_of(slocw), (unsigned long long)P.loc_w_t, cA * cF, 1);
    tdm_load_1d(lds_offset_of(sconv), (unsigned long long)P.loc_conv, cF * 2 * cKC, 2);
    __builtin_amdgcn_s_wait_tensorcnt(0);
  }
  __syncthreads();
#else
  for (int i = threadIdx.x; i < cA * cF; i += NTHR) slocw[i] = A.loc_w_t[i];
  for (int i = threadIdx.x; i < cF * 2 * cKC; i += NTHR) sconv[i] = A.loc_conv[i];
  __syncthreads();
#endif

  // one-time keys precompute: [B*TENC,1024] @ [1024,512] -> bf16 (both layouts)
  gemm_keys(A.enc, 1024, A.k_w_t, 1024, A.keys, A.keys_t, gw, lane);
  gsync(P.bar);

  for (int step = 0; step < cTMEL; ++step) {
    // attention GRU gates + (overlapped) location conv for this step
    gemm_bf(A.xcat, cXI, A.attn_wi_t, cXI, A.attn_bi, A.gi, cH3, nullptr, 0, cB, cH3, cXI, false, gw, lane);
    gemm_bf(A.xcat2, cX2, A.attn_wh_t, 512, A.attn_bh, A.gh, cH3, nullptr, 0, cB, cH3, 512, false, gw, lane);
    conv_phase(A, sconv, gtid);
    gsync(P.bar);
    gru_ew(A.gi, A.gh, A.h_attn, A.xcat2, cX2, gtid);      // -> new h_attn (f32 + bf16 in xcat2[:, :512])
    gsync(P.bar);
    gemm_bf(A.xcat2, cX2, A.q_w_t, 512, nullptr, A.qb, cA, nullptr, 0, cB, cA, 512, false, gw, lane);
    gsync(P.bar);
    score_phase(A, slocw, gw, lane);
    gsync(P.bar);
    softmax_ctx(A, step);
    gsync(P.bar);
    gemm_bf(A.xcat2, cX2, A.out_wi_t, cX2, A.out_bi, A.gi, cH3, nullptr, 0, cB, cH3, cX2, false, gw, lane);
    gemm_bf(A.xcat3, cX2, A.out_wh_t, 512, A.out_bh, A.gh, cH3, nullptr, 0, cB, cH3, 512, false, gw, lane);
    gsync(P.bar);
    gru_ew(A.gi, A.gh, A.h_out, A.xcat3, cX2, gtid);       // -> new h_out (f32 + bf16 in xcat3[:, :512])
    gsync(P.bar);
    mel_phase(A, step, gw, lane);                          // mel GEMM + masked outputs + gate
    gsync(P.bar);
    gemm_bf(A.melb, cM, A.pre_w1_t, cM, nullptr, nullptr, 0, A.h1b, 256, cB, 256, cM, true, gw, lane);
    gsync(P.bar);
    gemm_bf(A.h1b, 256, A.pre_w2_t, 256, nullptr, nullptr, 0, A.xcat, cXI, cB, 256, 256, true, gw, lane);
    gsync(P.bar);
  }
}

// ---------------- host launcher ----------------
extern "C" void kernel_launch(void* const* d_in, const int* in_sizes, int n_in,
                              void* d_out, int out_size, void* d_ws, size_t ws_size,
                              hipStream_t stream) {
  (void)in_sizes; (void)n_in; (void)out_size; (void)ws_size;
  Args A;
  A.enc = (const float*)d_in[0];
  A.text_lens = (const int*)d_in[2];
  A.mel_lens = (const int*)d_in[3];
  A.pre_w1 = (const float*)d_in[5];  A.pre_w2 = (const float*)d_in[6];
  A.attn_wi = (const float*)d_in[7]; A.attn_wh = (const float*)d_in[8];
  A.attn_bi = (const float*)d_in[9]; A.attn_bh = (const float*)d_in[10];
  A.q_w = (const float*)d_in[11];    A.k_w = (const float*)d_in[12];
  A.score_w = (const float*)d_in[13]; A.score_b = (const float*)d_in[14];
  A.loc_conv = (const float*)d_in[15]; A.loc_w = (const float*)d_in[16]; A.loc_b = (const float*)d_in[17];
  A.out_wi = (const float*)d_in[18]; A.out_wh = (const float*)d_in[19];
  A.out_bi = (const float*)d_in[20]; A.out_bh = (const float*)d_in[21];
  A.dec_w = (const float*)d_in[22];  A.dec_b = (const float*)d_in[23];
  A.gate_w = (const float*)d_in[24]; A.gate_b = (const float*)d_in[25];

  float* out = (float*)d_out;
  A.o_mel = out;                                    // [B,TMEL,M]
  A.o_gate = out + (size_t)cB * cTMEL * cM;         // [B,TMEL]
  A.o_attn = A.o_gate + (size_t)cB * cTMEL;         // [B,TMEL,TENC]
  A.o_mask = A.o_attn + (size_t)cB * cTMEL * cTENC; // [B,TMEL]

  char* ws = (char*)d_ws;
  size_t off = 0;
  auto alloc = [&](size_t bytes) -> void* {
    void* p = ws + off;
    off = (off + bytes + 255) & ~(size_t)255;
    return p;
  };
  A.bar       = (unsigned*)alloc(256);
  A.attn_wi_t = (u16*)alloc((size_t)cH3 * cXI * 2);
  A.attn_wh_t = (u16*)alloc((size_t)cH3 * 512 * 2);
  A.out_wi_t  = (u16*)alloc((size_t)cH3 * cX2 * 2);
  A.out_wh_t  = (u16*)alloc((size_t)cH3 * 512 * 2);
  A.q_w_t     = (u16*)alloc((size_t)cA * 512 * 2);
  A.k_w_t     = (u16*)alloc((size_t)cA * 1024 * 2);
  A.loc_w_t   = (u16*)alloc((size_t)cA * cF * 2);
  A.pre_w1_t  = (u16*)alloc((size_t)256 * 128 * 2);
  A.pre_w2_t  = (u16*)alloc((size_t)256 * 256 * 2);
  A.dec_w_t   = (u16*)alloc((size_t)cM * cX2 * 2);
  A.keys      = (u16*)alloc((size_t)cB * cTENC * cA * 2);
  A.keys_t    = (u16*)alloc((size_t)cB * cTENC * cA * 2);
  A.loc_f     = (u16*)alloc((size_t)cB * cTENC * cF * 2);
  A.xcat      = (u16*)alloc((size_t)cB * cXI * 2);
  A.xcat2     = (u16*)alloc((size_t)cB * cX2 * 2);
  A.xcat3     = (u16*)alloc((size_t)cB * cX2 * 2);
  A.h1b       = (u16*)alloc((size_t)cB * 256 * 2);
  A.melb      = (u16*)alloc((size_t)cB * cM * 2);
  A.h_attn    = (float*)alloc((size_t)cB * 512 * 4);
  A.h_out     = (float*)alloc((size_t)cB * 512 * 4);
  A.attw      = (float*)alloc((size_t)cB * cTENC * 4);
  A.attcum    = (float*)alloc((size_t)cB * cTENC * 4);
  A.qb        = (float*)alloc((size_t)cB * cA * 4);
  A.sb        = (float*)alloc((size_t)cB * cTENC * 4);
  A.gi        = (float*)alloc((size_t)cB * cH3 * 4);
  A.gh        = (float*)alloc((size_t)cB * cH3 * 4);

  (void)hipMemsetAsync(A.bar, 0, 256, stream);   // reset device-wide barrier (graph-capture safe)
  tacotron2_decoder_persistent<<<NBLK, NTHR, 0, stream>>>(A);
}
